// TransformerBlock_32066225832169
// MI455X (gfx1250) — matrix operand order repaired and checked by reading, not compiled, not run on hardware
//
#include <hip/hip_runtime.h>
#include <hip/hip_bf16.h>

// ---------------------------------------------------------------- constants
constexpr int Bv   = 8;
constexpr int Tv   = 4096;
constexpr int Dv   = 1024;
constexpr int Hv   = 16;
constexpr int DKv  = 64;
constexpr int CHv  = 64;
constexpr int NCv  = Tv / CHv;          // 64 chunks
constexpr int Ev   = 8;
constexpr int HIDv = 4096;
constexpr int NTOK = Bv * Tv;           // 32768
constexpr int CAPv = 5120;              // int(1.25 * NTOK / E)
constexpr int LDP  = 72;                // padded LDS row stride (16B-aligned, bank-spread)

typedef __attribute__((ext_vector_type(16))) __bf16 v16bf;
typedef __attribute__((ext_vector_type(8)))  float  v8f;

struct alignas(16) U4 { unsigned int x, y, z, w; };
union AFrag { v16bf v; U4 q[2]; };

// ---------------------------------------------------------------- async copy (gfx1250 ASYNCcnt path)
#if defined(__has_builtin)
#if __has_builtin(__builtin_amdgcn_global_load_async_to_lds_b128)
#define HAS_ASYNC_LDS 1
#endif
#endif
#ifndef HAS_ASYNC_LDS
#define HAS_ASYNC_LDS 0
#endif

typedef int v4i_vs __attribute__((vector_size(16)));
typedef __attribute__((address_space(1))) v4i_vs as1_v4i;
typedef __attribute__((address_space(3))) v4i_vs as3_v4i;

__device__ __forceinline__ void cp_g2l_b128(void* lds, const void* g) {
#if HAS_ASYNC_LDS
    __builtin_amdgcn_global_load_async_to_lds_b128(
        (as1_v4i*)g, (as3_v4i*)lds, 0, 0);
#else
    *(U4*)lds = *(const U4*)g;
#endif
}
__device__ __forceinline__ void wait_async_all() {
#if HAS_ASYNC_LDS
#if __has_builtin(__builtin_amdgcn_s_wait_asynccnt)
    __builtin_amdgcn_s_wait_asynccnt(0);
#else
    asm volatile("s_wait_asynccnt 0" ::: "memory");
#endif
#endif
}

// ---------------------------------------------------------------- helpers
__device__ __forceinline__ unsigned short f32_to_bf16(float f) {
    unsigned int u = __float_as_uint(f);
    unsigned int r = u + 0x7fffu + ((u >> 16) & 1u);   // round to nearest even
    return (unsigned short)(r >> 16);
}
__device__ __forceinline__ float bf2f(unsigned short h) {
    return __uint_as_float(((unsigned int)h) << 16);
}

// ---------------------------------------------------------------- weight transpose -> bf16
// src: f32 [K x N] row-major ; dst: bf16 [N x K] row-major
__global__ __launch_bounds__(256) void wtrans_kernel(const float* __restrict__ src,
                                                     unsigned short* __restrict__ dst,
                                                     int K, int N) {
    size_t gid = (size_t)blockIdx.x * 256 + threadIdx.x;
    if (gid >= (size_t)K * N) return;
    int k = (int)(gid / N);
    int n = (int)(gid % N);
    dst[(size_t)n * K + k] = f32_to_bf16(src[gid]);
}

// ---------------------------------------------------------------- rmsnorm -> bf16
__global__ __launch_bounds__(256) void rmsnorm_kernel(const float* __restrict__ x,
                                                      const float* __restrict__ scale,
                                                      unsigned short* __restrict__ out) {
    int row = blockIdx.x;
    int tid = threadIdx.x;
    __shared__ float red[256];
    const float* xr = x + (size_t)row * Dv;
    float s = 0.f;
    for (int d = tid; d < Dv; d += 256) { float v = xr[d]; s += v * v; }
    red[tid] = s;
    __syncthreads();
    for (int st = 128; st > 0; st >>= 1) {
        if (tid < st) red[tid] += red[tid + st];
        __syncthreads();
    }
    float inv = rsqrtf(red[0] / (float)Dv);
    for (int d = tid; d < Dv; d += 256)
        out[(size_t)row * Dv + d] = f32_to_bf16(xr[d] * inv * scale[d]);
}

// ---------------------------------------------------------------- WMMA bf16 GEMM
// C[M,N] = act( A_bf16[M,K] @ Bt_bf16[N,K]^T + bias ), out f32 and/or bf16
// grid: (M/64, N/128), block: 256 threads (8 waves, wave tile 32x32)
__global__ __launch_bounds__(256) void gemm_bf16_kernel(
    const unsigned short* __restrict__ A,
    const unsigned short* __restrict__ Bt,
    const float* __restrict__ bias,
    float* __restrict__ Cf,
    unsigned short* __restrict__ Cb,
    int M, int N, int K, int act_gelu)
{
    const int tid   = threadIdx.x;
    const int lane  = tid & 31;
    const int wid   = tid >> 5;
    const int waveM = wid & 1;
    const int waveN = wid >> 1;
    const int m0 = blockIdx.x * 64  + waveM * 32;
    const int n0 = blockIdx.y * 128 + waveN * 32;

    const int half = lane >> 4;
    const int lr   = lane & 15;

    v8f acc[2][2] = {};

    const unsigned short* pa0 = A  + (size_t)(m0 + lr) * K + half * 8;
    const unsigned short* pa1 = pa0 + (size_t)16 * K;
    const unsigned short* pb0 = Bt + (size_t)(n0 + lr) * K + half * 8;
    const unsigned short* pb1 = pb0 + (size_t)16 * K;

    for (int k0 = 0; k0 < K; k0 += 32) {
        AFrag a0, a1, b0, b1;
        a0.q[0] = *(const U4*)(pa0 + k0);
        a0.q[1] = *(const U4*)(pa0 + k0 + 16);
        a1.q[0] = *(const U4*)(pa1 + k0);
        a1.q[1] = *(const U4*)(pa1 + k0 + 16);
        b0.q[0] = *(const U4*)(pb0 + k0);
        b0.q[1] = *(const U4*)(pb0 + k0 + 16);
        b1.q[0] = *(const U4*)(pb1 + k0);
        b1.q[1] = *(const U4*)(pb1 + k0 + 16);
        __builtin_prefetch(pa0 + k0 + 32, 0, 0);
        __builtin_prefetch(pb0 + k0 + 32, 0, 0);
        acc[0][0] = __builtin_amdgcn_wmma_f32_16x16x32_bf16(false, a0.v, false, b0.v,
                        (short)0, acc[0][0], false, false);
        acc[0][1] = __builtin_amdgcn_wmma_f32_16x16x32_bf16(false, a0.v, false, b1.v,
                        (short)0, acc[0][1], false, false);
        acc[1][0] = __builtin_amdgcn_wmma_f32_16x16x32_bf16(false, a1.v, false, b0.v,
                        (short)0, acc[1][0], false, false);
        acc[1][1] = __builtin_amdgcn_wmma_f32_16x16x32_bf16(false, a1.v, false, b1.v,
                        (short)0, acc[1][1], false, false);
    }

    #pragma unroll
    for (int ti = 0; ti < 2; ++ti) {
        #pragma unroll
        for (int tj = 0; tj < 2; ++tj) {
            int col = n0 + tj * 16 + lr;
            float bval = bias ? bias[col] : 0.f;
            #pragma unroll
            for (int r = 0; r < 8; ++r) {
                int row = m0 + ti * 16 + half * 8 + r;
                float v = acc[ti][tj][r] + bval;
                if (act_gelu) v = 0.5f * v * (1.0f + erff(v * 0.70710678118f));
                size_t off = (size_t)row * N + col;
                if (Cf) Cf[off] = v;
                if (Cb) Cb[off] = f32_to_bf16(v);
            }
        }
    }
}

// ---------------------------------------------------------------- RoPE + phi epilogue
// y: f32 [NTOK x D] (token-major, col = h*64 + j)
// transposed==0 -> out[bh][t][dk] ; transposed==1 -> out[bh][dk][t]
__global__ __launch_bounds__(256) void rope_phi_kernel(const float* __restrict__ y,
                                                       unsigned short* __restrict__ out,
                                                       int transposed) {
    size_t gid = (size_t)blockIdx.x * 256 + threadIdx.x;   // over NTOK * D/2
    if (gid >= (size_t)NTOK * (Dv / 2)) return;
    int i   = (int)(gid & 31);          // 0..31 (rope half)
    int h   = (int)((gid >> 5) & 15);
    size_t tok = gid >> 9;              // 0..NTOK-1
    int t = (int)(tok & (Tv - 1));
    int b = (int)(tok >> 12);
    float x1 = y[tok * Dv + h * 64 + i];
    float x2 = y[tok * Dv + h * 64 + i + 32];
    float inv = powf(10000.0f, -(float)i / 32.0f);
    float ang = (float)t * inv;
    float sn = sinf(ang), cs = cosf(ang);
    float r1 = x1 * cs - x2 * sn;
    float r2 = x1 * sn + x2 * cs;
    // phi(x) = elu(x) + 1
    r1 = (r1 > 0.f) ? (r1 + 1.f) : __expf(r1);
    r2 = (r2 > 0.f) ? (r2 + 1.f) : __expf(r2);
    size_t bh = (size_t)b * Hv + h;
    if (!transposed) {
        out[(bh * Tv + t) * DKv + i]      = f32_to_bf16(r1);
        out[(bh * Tv + t) * DKv + i + 32] = f32_to_bf16(r2);
    } else {
        out[(bh * DKv + i) * Tv + t]        = f32_to_bf16(r1);
        out[(bh * DKv + i + 32) * Tv + t]   = f32_to_bf16(r2);
    }
}

// v: f32 [NTOK x D] -> vT bf16 [bh][dk][t]
__global__ __launch_bounds__(256) void vtrans_kernel(const float* __restrict__ y,
                                                     unsigned short* __restrict__ out) {
    size_t gid = (size_t)blockIdx.x * 256 + threadIdx.x;
    if (gid >= (size_t)NTOK * Dv) return;
    int col = (int)(gid & (Dv - 1));
    size_t tok = gid >> 10;
    int h = col >> 6, dk = col & 63;
    int t = (int)(tok & (Tv - 1));
    int b = (int)(tok >> 12);
    out[(((size_t)b * Hv + h) * DKv + dk) * Tv + t] = f32_to_bf16(y[gid]);
}

// g: f32 [NTOK x D] -> sigmoid -> bf16 [bh][t][dk]
__global__ __launch_bounds__(256) void sigmoid_g_kernel(const float* __restrict__ y,
                                                        unsigned short* __restrict__ out) {
    size_t gid = (size_t)blockIdx.x * 256 + threadIdx.x;
    if (gid >= (size_t)NTOK * Dv) return;
    int col = (int)(gid & (Dv - 1));
    size_t tok = gid >> 10;
    int h = col >> 6, dk = col & 63;
    int t = (int)(tok & (Tv - 1));
    int b = (int)(tok >> 12);
    float v = 1.0f / (1.0f + __expf(-y[gid]));
    out[(((size_t)b * Hv + h) * Tv + t) * DKv + dk] = f32_to_bf16(v);
}

// ---------------------------------------------------------------- linear attention scan (WMMA)
// one block (8 waves) per (b,h); sequential over 64 chunks.
// S (64x64) lives in per-wave f32 WMMA accumulators: wave w owns tiles 2w, 2w+1
// (tile tau: row-tile tau>>2, col-tile tau&3). Each chunk:
//   S += K^T V (WMMA, bf16 tiles from LDS) ; Z += sum_t K (scalar)
//   spill bf16 S^T to LDS ; den = q.Z (scalar)
//   num = Q S (WMMA via S^T B-fragments) ; out = g * num / den
__global__ __launch_bounds__(256) void attn_scan_kernel(
    const unsigned short* __restrict__ qb,   // [bh][t][dk]
    const unsigned short* __restrict__ kTb,  // [bh][dk][t]
    const unsigned short* __restrict__ vTb,  // [bh][dk][t]
    const unsigned short* __restrict__ gb,   // [bh][t][dk]
    unsigned short* __restrict__ attnout)    // [NTOK x D], col = h*64+e
{
    const int bh = blockIdx.x;
    const int b = bh / Hv, h = bh % Hv;
    const int tid  = threadIdx.x;
    const int lane = tid & 31;
    const int wid  = tid >> 5;
    const int half = lane >> 4;
    const int lr   = lane & 15;

    __shared__ unsigned short kt[DKv][LDP];   // [d][t]
    __shared__ unsigned short vt[DKv][LDP];   // [e][t]
    __shared__ unsigned short qt[CHv][LDP];   // [t][d]
    __shared__ unsigned short gt[CHv][LDP];   // [t][e]
    __shared__ unsigned short STb[DKv][LDP];  // S^T bf16: [e][d]
    __shared__ float Z[DKv];
    __shared__ float den[CHv];

    if (tid < DKv) Z[tid] = 0.f;
    __syncthreads();

    // wave-owned S tiles: tau0 = 2*wid, tau1 = 2*wid+1
    const int dt0 = (2 * wid) >> 2, et0 = (2 * wid) & 3;
    const int dt1 = (2 * wid + 1) >> 2, et1 = (2 * wid + 1) & 3;
    v8f sacc0 = {}, sacc1 = {};

    const size_t kbase = (size_t)bh * DKv * Tv;
    const size_t qbase = (size_t)bh * Tv * DKv;

    for (int c = 0; c < NCv; ++c) {
        const int tbase = c * CHv;
        // ---- stage tiles (async -> LDS when available) ----
        for (int p = tid; p < DKv * 8; p += 256) {
            int r = p >> 3, cb = (p & 7) * 8;
            cp_g2l_b128(&kt[r][cb], kTb + kbase + (size_t)r * Tv + tbase + cb);
            cp_g2l_b128(&vt[r][cb], vTb + kbase + (size_t)r * Tv + tbase + cb);
            cp_g2l_b128(&qt[r][cb], qb  + qbase + (size_t)(tbase + r) * DKv + cb);
            cp_g2l_b128(&gt[r][cb], gb  + qbase + (size_t)(tbase + r) * DKv + cb);
        }
        wait_async_all();
        __syncthreads();

        // ---- S += K^T V  (WMMA) ; Z += sum_t K (scalar lanes) ----
        #pragma unroll
        for (int k0 = 0; k0 < CHv; k0 += 32) {
            AFrag a0, a1, b0, b1;
            a0.q[0] = *(const U4*)(&kt[dt0 * 16 + lr][k0 + half * 8]);
            a0.q[1] = *(const U4*)(&kt[dt0 * 16 + lr][k0 + 16 + half * 8]);
            b0.q[0] = *(const U4*)(&vt[et0 * 16 + lr][k0 + half * 8]);
            b0.q[1] = *(const U4*)(&vt[et0 * 16 + lr][k0 + 16 + half * 8]);
            a1.q[0] = *(const U4*)(&kt[dt1 * 16 + lr][k0 + half * 8]);
            a1.q[1] = *(const U4*)(&kt[dt1 * 16 + lr][k0 + 16 + half * 8]);
            b1.q[0] = *(const U4*)(&vt[et1 * 16 + lr][k0 + half * 8]);
            b1.q[1] = *(const U4*)(&vt[et1 * 16 + lr][k0 + 16 + half * 8]);
            sacc0 = __builtin_amdgcn_wmma_f32_16x16x32_bf16(false, a0.v, false, b0.v,
                        (short)0, sacc0, false, false);
            sacc1 = __builtin_amdgcn_wmma_f32_16x16x32_bf16(false, a1.v, false, b1.v,
                        (short)0, sacc1, false, false);
        }
        if (tid < DKv) {
            float z = 0.f;
            for (int t = 0; t < CHv; ++t) z += bf2f(kt[tid][t]);
            Z[tid] += z;
        }
        __syncthreads();

        // ---- spill S^T (bf16) to LDS ; den[t] = q.Z + 1e-6 ----
        #pragma unroll
        for (int r = 0; r < 8; ++r) {
            STb[et0 * 16 + lr][dt0 * 16 + half * 8 + r] = f32_to_bf16(sacc0[r]);
            STb[et1 * 16 + lr][dt1 * 16 + half * 8 + r] = f32_to_bf16(sacc1[r]);
        }
        if (tid < CHv) {
            float ds = 0.f;
            for (int d = 0; d < DKv; ++d) ds += bf2f(qt[tid][d]) * Z[d];
            den[tid] = ds + 1e-6f;
        }
        __syncthreads();

        // ---- num = Q S  (WMMA, B-fragments from S^T) ; out = g*num/den ----
        const int tt0 = dt0, tt1 = dt1;       // same tile ownership for output
        v8f nacc0 = {}, nacc1 = {};
        #pragma unroll
        for (int k0 = 0; k0 < DKv; k0 += 32) {
            AFrag a0, a1, b0, b1;
            a0.q[0] = *(const U4*)(&qt[tt0 * 16 + lr][k0 + half * 8]);
            a0.q[1] = *(const U4*)(&qt[tt0 * 16 + lr][k0 + 16 + half * 8]);
            b0.q[0] = *(const U4*)(&STb[et0 * 16 + lr][k0 + half * 8]);
            b0.q[1] = *(const U4*)(&STb[et0 * 16 + lr][k0 + 16 + half * 8]);
            a1.q[0] = *(const U4*)(&qt[tt1 * 16 + lr][k0 + half * 8]);
            a1.q[1] = *(const U4*)(&qt[tt1 * 16 + lr][k0 + 16 + half * 8]);
            b1.q[0] = *(const U4*)(&STb[et1 * 16 + lr][k0 + half * 8]);
            b1.q[1] = *(const U4*)(&STb[et1 * 16 + lr][k0 + 16 + half * 8]);
            nacc0 = __builtin_amdgcn_wmma_f32_16x16x32_bf16(false, a0.v, false, b0.v,
                        (short)0, nacc0, false, false);
            nacc1 = __builtin_amdgcn_wmma_f32_16x16x32_bf16(false, a1.v, false, b1.v,
                        (short)0, nacc1, false, false);
        }
        #pragma unroll
        for (int r = 0; r < 8; ++r) {
            int t0r = tt0 * 16 + half * 8 + r;
            int e0c = et0 * 16 + lr;
            float o0 = bf2f(gt[t0r][e0c]) * (nacc0[r] / den[t0r]);
            attnout[((size_t)(b * Tv + tbase + t0r)) * Dv + h * DKv + e0c] = f32_to_bf16(o0);
            int t1r = tt1 * 16 + half * 8 + r;
            int e1c = et1 * 16 + lr;
            float o1 = bf2f(gt[t1r][e1c]) * (nacc1[r] / den[t1r]);
            attnout[((size_t)(b * Tv + tbase + t1r)) * Dv + h * DKv + e1c] = f32_to_bf16(o1);
        }
        __syncthreads();
    }
}

// ---------------------------------------------------------------- residual add
__global__ __launch_bounds__(256) void add_res_kernel(const float* __restrict__ x,
                                                      const float* __restrict__ t,
                                                      float* __restrict__ x1,
                                                      float* __restrict__ dout) {
    size_t gid = (size_t)blockIdx.x * 256 + threadIdx.x;
    if (gid >= (size_t)NTOK * Dv) return;
    float v = x[gid] + t[gid];
    x1[gid] = v;
    dout[gid] = v;
}

// ---------------------------------------------------------------- MoE router (wave per token)
__global__ __launch_bounds__(256) void router_kernel(
    const unsigned short* __restrict__ xn2,
    const float* __restrict__ rW,     // [D x E]
    const float* __restrict__ rb,     // [E]
    int* __restrict__ cnt,            // [E]
    float* __restrict__ importance,   // [E]
    int* __restrict__ tok_list,       // [E x CAP]
    float* __restrict__ gate_list)    // [E x CAP]
{
    int wave = (int)(((size_t)blockIdx.x * 256 + threadIdx.x) >> 5);
    int lane = threadIdx.x & 31;
    if (wave >= NTOK) return;
    size_t tok = (size_t)wave;
    float logit[Ev];
    #pragma unroll
    for (int e = 0; e < Ev; ++e) logit[e] = 0.f;
    for (int d = lane; d < Dv; d += 32) {
        float xv = bf2f(xn2[tok * Dv + d]);
        #pragma unroll
        for (int e = 0; e < Ev; ++e) logit[e] += xv * rW[(size_t)d * Ev + e];
    }
    #pragma unroll
    for (int e = 0; e < Ev; ++e)
        for (int off = 16; off > 0; off >>= 1)
            logit[e] += __shfl_xor(logit[e], off, 32);
    if (lane == 0) {
        float mx = -1e30f;
        for (int e = 0; e < Ev; ++e) { logit[e] += rb[e]; mx = fmaxf(mx, logit[e]); }
        float p[Ev], ps = 0.f;
        for (int e = 0; e < Ev; ++e) { p[e] = __expf(logit[e] - mx); ps += p[e]; }
        for (int e = 0; e < Ev; ++e) { p[e] /= ps; atomicAdd(&importance[e], p[e]); }
        int i1 = 0;
        for (int e = 1; e < Ev; ++e) if (p[e] > p[i1]) i1 = e;
        int i2 = (i1 == 0) ? 1 : 0;
        for (int e = 0; e < Ev; ++e) if (e != i1 && p[e] > p[i2]) i2 = e;
        float gs = p[i1] + p[i2];
        float g1 = p[i1] / gs, g2 = p[i2] / gs;
        int pos = atomicAdd(&cnt[i1], 1);
        if (pos < CAPv) { tok_list[i1 * CAPv + pos] = (int)tok; gate_list[i1 * CAPv + pos] = g1; }
        pos = atomicAdd(&cnt[i2], 1);
        if (pos < CAPv) { tok_list[i2 * CAPv + pos] = (int)tok; gate_list[i2 * CAPv + pos] = g2; }
    }
}

// ---------------------------------------------------------------- MoE gather / scatter
__global__ __launch_bounds__(256) void gather_kernel(const unsigned short* __restrict__ xn2,
                                                     const int* __restrict__ cnt,
                                                     const int* __restrict__ tok_list,
                                                     unsigned short* __restrict__ Xg, int e) {
    int r = blockIdx.x, tid = threadIdx.x;
    int n = min(cnt[e], CAPv);
    if (r < n) {
        size_t tok = (size_t)tok_list[e * CAPv + r];
        for (int d = tid; d < Dv; d += 256)
            Xg[(size_t)r * Dv + d] = xn2[tok * Dv + d];
    } else {
        for (int d = tid; d < Dv; d += 256)
            Xg[(size_t)r * Dv + d] = 0;
    }
}

__global__ __launch_bounds__(256) void scatter_kernel(const float* __restrict__ Og,
                                                      const int* __restrict__ cnt,
                                                      const int* __restrict__ tok_list,
                                                      const float* __restrict__ gate_list,
                                                      float* __restrict__ dout, int e) {
    size_t gid = (size_t)blockIdx.x * 256 + threadIdx.x;
    if (gid >= (size_t)CAPv * Dv) return;
    int r = (int)(gid / Dv);
    int d = (int)(gid % Dv);
    int n = min(cnt[e], CAPv);
    if (r >= n) return;
    size_t tok = (size_t)tok_list[e * CAPv + r];
    float g = gate_list[e * CAPv + r];
    atomicAdd(&dout[tok * Dv + d], g * Og[gid]);
}

// ---------------------------------------------------------------- aux loss
__global__ void aux_kernel(const float* __restrict__ importance,
                           const int* __restrict__ cnt,
                           float* __restrict__ dout_aux) {
    if (threadIdx.x == 0) {
        float imps = 0.f, lds = 0.f, ax = 0.f;
        float imp[Ev], ld[Ev];
        for (int e = 0; e < Ev; ++e) {
            imp[e] = importance[e];
            ld[e]  = (float)min(cnt[e], CAPv);
            imps += imp[e]; lds += ld[e];
        }
        for (int e = 0; e < Ev; ++e) ax += (imp[e] / imps) * (ld[e] / lds);
        *dout_aux = ax * (float)(Ev * Ev);
    }
}

// ================================================================ host
extern "C" void kernel_launch(void* const* d_in, const int* in_sizes, int n_in,
                              void* d_out, int out_size, void* d_ws, size_t ws_size,
                              hipStream_t stream) {
    (void)in_sizes; (void)n_in; (void)out_size; (void)ws_size;
    const float* x   = (const float*)d_in[0];
    const float* n1s = (const float*)d_in[1];
    const float* n2s = (const float*)d_in[2];
    const float* Wq  = (const float*)d_in[3];
    const float* bq  = (const float*)d_in[4];
    const float* Wk  = (const float*)d_in[5];
    const float* bk  = (const float*)d_in[6];
    const float* Wv  = (const float*)d_in[7];
    const float* bv  = (const float*)d_in[8];
    const float* Wg  = (const float*)d_in[9];
    const float* bg  = (const float*)d_in[10];
    const float* Wo  = (const float*)d_in[11];
    const float* bo  = (const float*)d_in[12];
    const float* rW  = (const float*)d_in[13];
    const float* rb  = (const float*)d_in[14];
    const float* W1  = (const float*)d_in[15];
    const float* b1  = (const float*)d_in[16];
    const float* W2  = (const float*)d_in[17];
    const float* b2  = (const float*)d_in[18];
    float* dout = (float*)d_out;

    char* ws = (char*)d_ws;
    size_t off = 0;
    auto alloc = [&](size_t bytes) -> void* {
        off = (off + 255) & ~(size_t)255;
        void* p = ws + off;
        off += bytes;
        return p;
    };
    const size_t ND = (size_t)NTOK * Dv;

    unsigned short* wqT = (unsigned short*)alloc((size_t)Dv * Dv * 2);
    unsigned short* wkT = (unsigned short*)alloc((size_t)Dv * Dv * 2);
    unsigned short* wvT = (unsigned short*)alloc((size_t)Dv * Dv * 2);
    unsigned short* wgT = (unsigned short*)alloc((size_t)Dv * Dv * 2);
    unsigned short* woT = (unsigned short*)alloc((size_t)Dv * Dv * 2);
    unsigned short* w1T = (unsigned short*)alloc((size_t)Ev * HIDv * Dv * 2);
    unsigned short* w2T = (unsigned short*)alloc((size_t)Ev * Dv * HIDv * 2);
    unsigned short* xn1 = (unsigned short*)alloc(ND * 2);
    float*          tmp = (float*)alloc(ND * 4);
    unsigned short* qb  = (unsigned short*)alloc(ND * 2);
    unsigned short* kTb = (unsigned short*)alloc(ND * 2);
    unsigned short* vTb = (unsigned short*)alloc(ND * 2);
    unsigned short* gb  = (unsigned short*)alloc(ND * 2);
    unsigned short* ao  = (unsigned short*)alloc(ND * 2);
    float*          x1  = (float*)alloc(ND * 4);
    unsigned short* xn2 = (unsigned short*)alloc(ND * 2);
    int*   cnt       = (int*)alloc(Ev * 4);
    float* importance= (float*)alloc(Ev * 4);
    int*   tok_list  = (int*)alloc((size_t)Ev * CAPv * 4);
    float* gate_list = (float*)alloc((size_t)Ev * CAPv * 4);
    unsigned short* Xg = (unsigned short*)alloc((size_t)CAPv * Dv * 2);
    unsigned short* Hg = (unsigned short*)alloc((size_t)CAPv * HIDv * 2);
    float*          Og = (float*)alloc((size_t)CAPv * Dv * 4);

    (void)hipMemsetAsync(cnt, 0, Ev * 4, stream);
    (void)hipMemsetAsync(importance, 0, Ev * 4, stream);

    // --- weight transposes (f32 -> bf16 [N x K]) ---
    {
        int nblk = (Dv * Dv + 255) / 256;
        wtrans_kernel<<<nblk, 256, 0, stream>>>(Wq, wqT, Dv, Dv);
        wtrans_kernel<<<nblk, 256, 0, stream>>>(Wk, wkT, Dv, Dv);
        wtrans_kernel<<<nblk, 256, 0, stream>>>(Wv, wvT, Dv, Dv);
        wtrans_kernel<<<nblk, 256, 0, stream>>>(Wg, wgT, Dv, Dv);
        wtrans_kernel<<<nblk, 256, 0, stream>>>(Wo, woT, Dv, Dv);
        int nblk1 = (Dv * HIDv + 255) / 256;
        for (int e = 0; e < Ev; ++e) {
            wtrans_kernel<<<nblk1, 256, 0, stream>>>(W1 + (size_t)e * Dv * HIDv,
                                                     w1T + (size_t)e * HIDv * Dv, Dv, HIDv);
            wtrans_kernel<<<nblk1, 256, 0, stream>>>(W2 + (size_t)e * HIDv * Dv,
                                                     w2T + (size_t)e * Dv * HIDv, HIDv, Dv);
        }
    }

    // --- attention path ---
    rmsnorm_kernel<<<NTOK, 256, 0, stream>>>(x, n1s, xn1);

    dim3 gproj(NTOK / 64, Dv / 128);
    size_t half_elems = ND / 2;
    int rope_blk = (int)((half_elems + 255) / 256);
    int full_blk = (int)((ND + 255) / 256);

    gemm_bf16_kernel<<<gproj, 256, 0, stream>>>(xn1, wqT, bq, tmp, nullptr, NTOK, Dv, Dv, 0);
    rope_phi_kernel<<<rope_blk, 256, 0, stream>>>(tmp, qb, 0);

    gemm_bf16_kernel<<<gproj, 256, 0, stream>>>(xn1, wkT, bk, tmp, nullptr, NTOK, Dv, Dv, 0);
    rope_phi_kernel<<<rope_blk, 256, 0, stream>>>(tmp, kTb, 1);

    gemm_bf16_kernel<<<gproj, 256, 0, stream>>>(xn1, wvT, bv, tmp, nullptr, NTOK, Dv, Dv, 0);
    vtrans_kernel<<<full_blk, 256, 0, stream>>>(tmp, vTb);

    gemm_bf16_kernel<<<gproj, 256, 0, stream>>>(xn1, wgT, bg, tmp, nullptr, NTOK, Dv, Dv, 0);
    sigmoid_g_kernel<<<full_blk, 256, 0, stream>>>(tmp, gb);

    attn_scan_kernel<<<Bv * Hv, 256, 0, stream>>>(qb, kTb, vTb, gb, ao);

    gemm_bf16_kernel<<<gproj, 256, 0, stream>>>(ao, woT, bo, tmp, nullptr, NTOK, Dv, Dv, 0);
    add_res_kernel<<<full_blk, 256, 0, stream>>>(x, tmp, x1, dout);

    // --- MoE path ---
    rmsnorm_kernel<<<NTOK, 256, 0, stream>>>(x1, n2s, xn2);
    router_kernel<<<NTOK / 8, 256, 0, stream>>>(xn2, rW, rb, cnt, importance, tok_list, gate_list);

    dim3 g1(CAPv / 64, HIDv / 128);
    dim3 g2(CAPv / 64, Dv / 128);
    int scat_blk = (int)(((size_t)CAPv * Dv + 255) / 256);
    for (int e = 0; e < Ev; ++e) {
        gather_kernel<<<CAPv, 256, 0, stream>>>(xn2, cnt, tok_list, Xg, e);
        gemm_bf16_kernel<<<g1, 256, 0, stream>>>(Xg, w1T + (size_t)e * HIDv * Dv,
                                                 b1 + (size_t)e * HIDv,
                                                 nullptr, Hg, CAPv, HIDv, Dv, 1);
        gemm_bf16_kernel<<<g2, 256, 0, stream>>>(Hg, w2T + (size_t)e * Dv * HIDv,
                                                 b2 + (size_t)e * Dv,
                                                 Og, nullptr, CAPv, Dv, HIDv, 0);
        scatter_kernel<<<scat_blk, 256, 0, stream>>>(Og, cnt, tok_list, gate_list, dout, e);
    }

    aux_kernel<<<1, 32, 0, stream>>>(importance, cnt, dout + ND);
}
